// SimpleGateLoopLayer_13159779795493
// MI455X (gfx1250) — matrix operand order repaired and checked by reading, not compiled, not run on hardware
//
#include <hip/hip_runtime.h>
#include <hip/hip_bf16.h>

// ---------------------------------------------------------------------------
// Problem constants (B=4, N=4096, D=1024)
// ---------------------------------------------------------------------------
#define BATCH   4
#define SEQN    4096
#define D_DIM   1024
#define E_DIM   3072            // 3*D
#define ROWS    (BATCH * SEQN)  // 16384
#define NCHUNK  16
#define CHUNK   (SEQN / NCHUNK) // 256

typedef __attribute__((ext_vector_type(16))) __bf16 v16bf;
typedef __attribute__((ext_vector_type(8)))  float  v8f;

union FragBF {
    uint4 u[2];   // 2 x 16B = 8 VGPRs of packed bf16
    v16bf v;
};

static __device__ __forceinline__ unsigned short f2bf(float f) {
    unsigned int u = __float_as_uint(f);
    unsigned int r = u + 0x7FFFu + ((u >> 16) & 1u);  // round-to-nearest-even
    return (unsigned short)(r >> 16);
}

// raw 32-bit LDS offset for async global->LDS instructions
typedef __attribute__((address_space(3))) const void lds_cv;
static __device__ __forceinline__ unsigned lds_addr_of(const void* p) {
    return (unsigned)(unsigned long long)(lds_cv*)p;
}

// gfx1250 direct async global->LDS 128-bit copy (ASYNCcnt-tracked)
#define ASYNC_COPY_B128(ldsoff, gptr)                                      \
    asm volatile("global_load_async_to_lds_b128 %0, %1, off"               \
                 :: "v"(ldsoff),                                           \
                    "v"((unsigned long long)(size_t)(gptr)) : "memory")

// ---------------------------------------------------------------------------
// Kernel 1: RMSNorm(x) * sqrt(D) * gamma  ->  bf16 row  (one block per row)
// ---------------------------------------------------------------------------
__global__ __launch_bounds__(256) void rmsnorm_cast_kernel(
    const float* __restrict__ x, const float* __restrict__ gamma,
    unsigned short* __restrict__ xn)
{
    const int row = blockIdx.x;
    const int tid = threadIdx.x;
    const size_t base = (size_t)row * D_DIM;

    float ss = 0.f;
    float vals[D_DIM / 256];
#pragma unroll
    for (int i = 0; i < D_DIM / 256; ++i) {
        float v = x[base + tid + i * 256];
        vals[i] = v;
        ss += v * v;
    }
#pragma unroll
    for (int off = 16; off > 0; off >>= 1)
        ss += __shfl_xor(ss, off, 32);

    __shared__ float wsum[8];
    const int lane = tid & 31, wv = tid >> 5;
    if (lane == 0) wsum[wv] = ss;
    __syncthreads();
    if (tid == 0) {
        float t = 0.f;
#pragma unroll
        for (int i = 0; i < 8; ++i) t += wsum[i];
        wsum[0] = t;
    }
    __syncthreads();
    const float norm  = sqrtf(wsum[0]);
    const float scale = sqrtf((float)D_DIM) / fmaxf(norm, 1e-12f);

#pragma unroll
    for (int i = 0; i < D_DIM / 256; ++i) {
        int c = tid + i * 256;
        xn[base + c] = f2bf(vals[i] * scale * gamma[c]);
    }
}

// ---------------------------------------------------------------------------
// Kernel 2: cast W (3072x1024 fp32) -> bf16
// ---------------------------------------------------------------------------
__global__ __launch_bounds__(256) void wcast_kernel(
    const float* __restrict__ w, unsigned short* __restrict__ wb, int n)
{
    for (int i = blockIdx.x * blockDim.x + threadIdx.x; i < n;
         i += gridDim.x * blockDim.x)
        wb[i] = f2bf(w[i]);
}

// ---------------------------------------------------------------------------
// Kernel 3: WMMA GEMM  qkva[row][e] = sum_k xn[row][k] * W[e][k]
//   block tile 128(M) x 128(N), K-step 32, fully unrolled K pipeline
//   8 waves (4 along M x 2 along N), each owns 32x64 = 2x4 WMMA tiles
//   double-buffered LDS fed by async global->LDS b128 copies
// ---------------------------------------------------------------------------
#define LDS_STRIDE 40   // 32 + 8 pad (ushort units) to spread LDS banks
#define STAGE_USH  (128 * LDS_STRIDE)
#define KSTAGES    (D_DIM / 32)   // 32

__global__ __launch_bounds__(256) void qkva_gemm_kernel(
    const unsigned short* __restrict__ Xn, const unsigned short* __restrict__ Wb,
    float* __restrict__ qbuf, float* __restrict__ kvbuf, float* __restrict__ abuf)
{
    __shared__ unsigned short sA[2][STAGE_USH];  // 128 rows x 32 bf16 (M x K)
    __shared__ unsigned short sB[2][STAGE_USH];  // 128 rows x 32 bf16 (N x K)

    const int tid  = threadIdx.x;
    const int lane = tid & 31;
    const int wv   = tid >> 5;
    const int wm   = wv & 3;        // 4 waves along M (32 rows each)
    const int wn   = wv >> 2;       // 2 waves along N (64 cols each)
    const int half = lane >> 4;     // lane half selects K sub-range
    const int mr   = lane & 15;     // M-row (A) / N-col (B,C) within tile

    const int n0 = blockIdx.x * 128;
    const int m0 = blockIdx.y * 128;

    v8f acc[2][4];
#pragma unroll
    for (int i = 0; i < 2; ++i)
#pragma unroll
        for (int j = 0; j < 4; ++j)
#pragma unroll
            for (int e = 0; e < 8; ++e) acc[i][j][e] = 0.f;

    // ---- staging geometry: 256 threads fill 128x32 bf16 per matrix ----
    const int sr = tid >> 2;            // 0..63  (rows sr and sr+64)
    const int sc = (tid & 3) * 8;       // 0,8,16,24 (bf16 elems; 16B chunks)

    const unsigned ldsA0 = lds_addr_of(&sA[0][sr * LDS_STRIDE + sc]);
    const unsigned ldsA1 = lds_addr_of(&sA[0][(sr + 64) * LDS_STRIDE + sc]);
    const unsigned ldsB0 = lds_addr_of(&sB[0][sr * LDS_STRIDE + sc]);
    const unsigned ldsB1 = lds_addr_of(&sB[0][(sr + 64) * LDS_STRIDE + sc]);
    const unsigned stageBytes = STAGE_USH * 2;

    const unsigned short* gA0 = Xn + (size_t)(m0 + sr) * D_DIM + sc;
    const unsigned short* gA1 = Xn + (size_t)(m0 + sr + 64) * D_DIM + sc;
    const unsigned short* gB0 = Wb + (size_t)(n0 + sr) * D_DIM + sc;
    const unsigned short* gB1 = Wb + (size_t)(n0 + sr + 64) * D_DIM + sc;

    // prologue: fill stage 0 (k0 = 0)
    ASYNC_COPY_B128(ldsA0, gA0);
    ASYNC_COPY_B128(ldsA1, gA1);
    ASYNC_COPY_B128(ldsB0, gB0);
    ASYNC_COPY_B128(ldsB1, gB1);

#pragma unroll
    for (int kt = 0; kt < KSTAGES; ++kt) {
        const int cur = kt & 1;
        // prefetch next K-stage into the other buffer (freed by the barrier
        // at the end of iteration kt-1)
        if (kt < KSTAGES - 1) {
            const unsigned boff = (cur ^ 1) * stageBytes;
            const int knext = (kt + 1) * 32;
            ASYNC_COPY_B128(ldsA0 + boff, gA0 + knext);
            ASYNC_COPY_B128(ldsA1 + boff, gA1 + knext);
            ASYNC_COPY_B128(ldsB0 + boff, gB0 + knext);
            ASYNC_COPY_B128(ldsB1 + boff, gB1 + knext);
            asm volatile("s_wait_asynccnt 0x4" ::: "memory");  // cur stage done
        } else {
            asm volatile("s_wait_asynccnt 0x0" ::: "memory");
        }
        __syncthreads();   // cur-stage LDS data visible to all waves

        // A fragments: ISA 16-bit A 16x32 layout:
        //   VGPR0-3: K = half*8 .. +7 ; VGPR4-7: K = 16 + half*8 .. +7
        FragBF fa[2], fb[4];
#pragma unroll
        for (int tm = 0; tm < 2; ++tm) {
            const int arow = (wm * 32 + tm * 16 + mr) * LDS_STRIDE;
            fa[tm].u[0] = *(const uint4*)&sA[cur][arow + half * 8];
            fa[tm].u[1] = *(const uint4*)&sA[cur][arow + 16 + half * 8];
        }
        // B fragments: same K layout as A (VGPR0-3: K = half*8 .. +7 ; VGPR4-7: K = 16 + half*8 .. +7)
#pragma unroll
        for (int tn = 0; tn < 4; ++tn) {
            const int brow = (wn * 64 + tn * 16 + mr) * LDS_STRIDE;
            fb[tn].u[0] = *(const uint4*)&sB[cur][brow + half * 8];
            fb[tn].u[1] = *(const uint4*)&sB[cur][brow + 16 + half * 8];
        }
#pragma unroll
        for (int tm = 0; tm < 2; ++tm)
#pragma unroll
            for (int tn = 0; tn < 4; ++tn)
                acc[tm][tn] = __builtin_amdgcn_wmma_f32_16x16x32_bf16(
                    false, fa[tm].v, false, fb[tn].v,
                    (short)0, acc[tm][tn], false, false);
        __syncthreads();   // all waves done reading cur buffer
    }

    // epilogue: C layout -> VGPR r holds M = r + half*8, N = lane&15
    const int seg = n0 >> 10;          // uniform per block (128 | 1024)
    const int lc0 = (n0 & 1023) + wn * 64 + mr;
#pragma unroll
    for (int tm = 0; tm < 2; ++tm) {
        const int gm = m0 + wm * 32 + tm * 16 + half * 8;
#pragma unroll
        for (int tn = 0; tn < 4; ++tn) {
            const int lc = lc0 + tn * 16;
#pragma unroll
            for (int r = 0; r < 8; ++r) {
                const size_t off = (size_t)(gm + r) * D_DIM + lc;
                const float  v   = acc[tm][tn][r];
                if (seg == 0)      qbuf[off]  = v;
                else if (seg == 1) kvbuf[off] = v;
                else               abuf[off]  = 1.f / (1.f + __expf(-v));
            }
        }
    }
}

// ---------------------------------------------------------------------------
// Chunked gated scan: state_t = a_t * state_{t-1} + kv_t along N
// ---------------------------------------------------------------------------
__global__ __launch_bounds__(256) void scan_partial_kernel(
    const float* __restrict__ kvbuf, const float* __restrict__ abuf,
    float* __restrict__ cA, float* __restrict__ cS)
{
    const int tid = blockIdx.x * blockDim.x + threadIdx.x; // B*NCHUNK*D threads
    const int d = tid & (D_DIM - 1);
    const int c = (tid >> 10) & (NCHUNK - 1);
    const int b = tid >> 14;
    size_t base = ((size_t)(b * SEQN + c * CHUNK)) * D_DIM + d;

    float A = 1.f, S = 0.f;
    for (int i = 0; i < CHUNK; ++i) {
        const float a  = abuf[base];
        const float kv = kvbuf[base];
        S = fmaf(a, S, kv);
        A *= a;
        base += D_DIM;
    }
    cA[tid] = A;   // index == (b*NCHUNK + c)*D + d
    cS[tid] = S;
}

__global__ __launch_bounds__(256) void scan_carry_kernel(
    const float* __restrict__ cA, const float* __restrict__ cS,
    float* __restrict__ carry)
{
    const int tid = blockIdx.x * blockDim.x + threadIdx.x; // B*D threads
    const int d = tid & (D_DIM - 1);
    const int b = tid >> 10;
    float state = 0.f;
    for (int c = 0; c < NCHUNK; ++c) {
        const int idx = (b * NCHUNK + c) * D_DIM + d;
        carry[idx] = state;                       // state entering chunk c
        state = fmaf(cA[idx], state, cS[idx]);
    }
}

__global__ __launch_bounds__(256) void scan_final_kernel(
    const float* __restrict__ qbuf, const float* __restrict__ kvbuf,
    const float* __restrict__ abuf, const float* __restrict__ carry,
    float* __restrict__ out)
{
    const int tid = blockIdx.x * blockDim.x + threadIdx.x;
    const int d = tid & (D_DIM - 1);
    const int c = (tid >> 10) & (NCHUNK - 1);
    const int b = tid >> 14;
    size_t base = ((size_t)(b * SEQN + c * CHUNK)) * D_DIM + d;

    float state = carry[tid];
    for (int i = 0; i < CHUNK; ++i) {
        const float a  = abuf[base];
        const float kv = kvbuf[base];
        state = fmaf(a, state, kv);
        out[base] = qbuf[base] * state;
        base += D_DIM;
    }
}

// ---------------------------------------------------------------------------
// Host-side launcher
// ---------------------------------------------------------------------------
extern "C" void kernel_launch(void* const* d_in, const int* in_sizes, int n_in,
                              void* d_out, int out_size, void* d_ws, size_t ws_size,
                              hipStream_t stream) {
    const float* x     = (const float*)d_in[0];   // [4,4096,1024]
    const float* W     = (const float*)d_in[1];   // [3072,1024]
    const float* gamma = (const float*)d_in[2];   // [1024]
    float* out = (float*)d_out;

    // workspace layout
    char* p = (char*)d_ws;
    unsigned short* Xn = (unsigned short*)p;  p += (size_t)ROWS * D_DIM * 2;   // 32 MB
    unsigned short* Wb = (unsigned short*)p;  p += (size_t)E_DIM * D_DIM * 2;  //  6 MB
    float* qbuf  = (float*)p;  p += (size_t)ROWS * D_DIM * 4;                  // 64 MB
    float* kvbuf = (float*)p;  p += (size_t)ROWS * D_DIM * 4;                  // 64 MB
    float* abuf  = (float*)p;  p += (size_t)ROWS * D_DIM * 4;                  // 64 MB
    float* cA    = (float*)p;  p += (size_t)BATCH * NCHUNK * D_DIM * 4;
    float* cS    = (float*)p;  p += (size_t)BATCH * NCHUNK * D_DIM * 4;
    float* carry = (float*)p;  p += (size_t)BATCH * NCHUNK * D_DIM * 4;
    (void)ws_size; (void)in_sizes; (void)n_in; (void)out_size;

    rmsnorm_cast_kernel<<<ROWS, 256, 0, stream>>>(x, gamma, Xn);
    wcast_kernel<<<2048, 256, 0, stream>>>(W, Wb, E_DIM * D_DIM);

    dim3 ggrid(E_DIM / 128, ROWS / 128);  // (24, 128)
    qkva_gemm_kernel<<<ggrid, 256, 0, stream>>>(Xn, Wb, qbuf, kvbuf, abuf);

    scan_partial_kernel<<<(BATCH * NCHUNK * D_DIM) / 256, 256, 0, stream>>>(kvbuf, abuf, cA, cS);
    scan_carry_kernel<<<(BATCH * D_DIM) / 256, 256, 0, stream>>>(cA, cS, carry);
    scan_final_kernel<<<(BATCH * NCHUNK * D_DIM) / 256, 256, 0, stream>>>(qbuf, kvbuf, abuf, carry, out);
}
